// InvariantPointAttention_60730837566346
// MI455X (gfx1250) — matrix operand order repaired and checked by reading, not compiled, not run on hardware
//
#include <hip/hip_runtime.h>
#include <hip/hip_bf16.h>

// ---------------- CDNA5 WMMA types / helpers ----------------
typedef __attribute__((ext_vector_type(16))) __bf16 v16bf;
typedef __attribute__((ext_vector_type(8)))  float  v8f;
typedef __attribute__((ext_vector_type(4)))  float  f4;

#define Nres 1024
#define Cdim 384
#define Hh   12
#define CH   32
#define Pp   4
#define PD   144
#define DPAIR 128
#define SCALE 0.17677669529663687f   /* 32^-0.5 */
#define NSPLIT 4

union F16x16 { v16bf v; unsigned u[8]; };

// pack two f32 into packed bf16 (truncate) with one v_perm_b32
__device__ __forceinline__ unsigned pack2(float lo, float hi) {
    union { float f; unsigned u; } a, b;
    a.f = lo; b.f = hi;
    return __builtin_amdgcn_perm(b.u /*src0=hi*/, a.u /*src1=lo*/, 0x07060302u);
}

__device__ __forceinline__ __bf16 f2bf(float x) {
    union { float f; unsigned u; } v; v.f = x;
    unsigned short h = (unsigned short)(v.u >> 16);
    __bf16 o; __builtin_memcpy(&o, &h, 2);
    return o;
}

__device__ __forceinline__ v8f wmma_bf16(v16bf a, v16bf b, v8f c) {
    return __builtin_amdgcn_wmma_f32_16x16x32_bf16(false, a, false, b,
                                                   (short)0, c, false, false);
}

// 16x32 bf16 A-fragment from f32 rows (LDS / cached paths).
// Per-lane K-pattern: contiguous run [16*kh, +16).
__device__ __forceinline__ v16bf a_frag_f32(const float* row, int kh) {
    float r0[8], r1[8];
    __builtin_memcpy(r0, row + 16 * kh, 32);
    __builtin_memcpy(r1, row + 16 * kh + 8, 32);
    F16x16 f;
#pragma unroll
    for (int j = 0; j < 4; ++j) {
        f.u[j]     = pack2(r0[2 * j], r0[2 * j + 1]);
        f.u[4 + j] = pack2(r1[2 * j], r1[2 * j + 1]);
    }
    return f.v;
}

// Same, but with non-temporal (streaming) loads for the 512MB pair stream:
// read-once data must not evict L2-resident k/v tiles.
__device__ __forceinline__ v16bf a_frag_f32_nt(const float* row, int kh) {
    const f4* p0 = (const f4*)(row + 16 * kh);
    const f4* p1 = (const f4*)(row + 16 * kh + 8);
    f4 r0 = __builtin_nontemporal_load(p0);
    f4 r1 = __builtin_nontemporal_load(p0 + 1);
    f4 r2 = __builtin_nontemporal_load(p1);
    f4 r3 = __builtin_nontemporal_load(p1 + 1);
    F16x16 f;
    f.u[0] = pack2(r0.x, r0.y); f.u[1] = pack2(r0.z, r0.w);
    f.u[2] = pack2(r1.x, r1.y); f.u[3] = pack2(r1.z, r1.w);
    f.u[4] = pack2(r2.x, r2.y); f.u[5] = pack2(r2.z, r2.w);
    f.u[6] = pack2(r3.x, r3.y); f.u[7] = pack2(r3.z, r3.w);
    return f.v;
}

// async memory -> LDS, 16 bytes per lane (GVS addressing), ASYNCcnt-tracked
__device__ __forceinline__ void async_b128(unsigned lds_off, unsigned byte_off,
                                           const void* saddr) {
    asm volatile("global_load_async_to_lds_b128 %0, %1, %2"
                 :: "v"(lds_off), "v"(byte_off), "s"(saddr) : "memory");
}

// ---------------- Kernel A: projections + rotations + sk ----------------
__global__ __launch_bounds__(128) void ipa_proj(
    const float* __restrict__ single, const float* __restrict__ rot,
    const float* __restrict__ Wq, const float* __restrict__ bq,
    const float* __restrict__ Wk, const float* __restrict__ bk,
    const float* __restrict__ Wv, const float* __restrict__ bv,
    const float* __restrict__ Wpq, const float* __restrict__ bpq,
    const float* __restrict__ Wpk, const float* __restrict__ bpk,
    __bf16* __restrict__ qb, __bf16* __restrict__ kb, __bf16* __restrict__ vb,
    __bf16* __restrict__ qgb, __bf16* __restrict__ kgb, float* __restrict__ skw)
{
    __shared__ float sX[16 * Cdim];
    __shared__ float pts[16 * 288];    // raw q_pts (0..143) / k_pts (144..287)

    const int tid = threadIdx.x, lane = tid & 31, w = tid >> 5;
    const int nbase = blockIdx.x * 16;
    const int kh = lane >> 4, lc = lane & 15;

    for (int i = tid; i < 16 * Cdim; i += 128)
        sX[i] = single[(size_t)nbase * Cdim + i];
    __syncthreads();

    for (int job = w; job < 90; job += 4) {
        const float *W, *bias; int outdim, colbase, kind;
        if (job < 24)      { W = Wq;  bias = bq;  outdim = Cdim; colbase = job * 16;        kind = 0; }
        else if (job < 48) { W = Wk;  bias = bk;  outdim = Cdim; colbase = (job - 24) * 16; kind = 1; }
        else if (job < 72) { W = Wv;  bias = bv;  outdim = Cdim; colbase = (job - 48) * 16; kind = 2; }
        else if (job < 81) { W = Wpq; bias = bpq; outdim = PD;   colbase = (job - 72) * 16; kind = 3; }
        else               { W = Wpk; bias = bpk; outdim = PD;   colbase = (job - 81) * 16; kind = 4; }

        v8f acc = {};
        for (int ks = 0; ks < 12; ++ks) {
            v16bf af = a_frag_f32(&sX[lc * Cdim + 32 * ks], kh);
            F16x16 bfm;
#pragma unroll
            for (int j = 0; j < 8; ++j) {
                int Kb = 2 * j + 16 * kh + 32 * ks;
                float lo = W[(size_t)Kb * outdim + colbase + lc];
                float hi = W[(size_t)(Kb + 1) * outdim + colbase + lc];
                bfm.u[j] = pack2(lo, hi);
            }
            acc = wmma_bf16(af, bfm.v, acc);
        }
        float bval = bias[colbase + lc];
        if (kind <= 2) {
            __bf16* dst = (kind == 0) ? qb : (kind == 1) ? kb : vb;
#pragma unroll
            for (int i = 0; i < 8; ++i) {
                int row = i + 8 * kh;
                dst[(size_t)(nbase + row) * Cdim + colbase + lc] = f2bf(acc[i] + bval);
            }
        } else {
            float* pdst = pts + ((kind == 3) ? 0 : 144);
#pragma unroll
            for (int i = 0; i < 8; ++i) {
                int row = i + 8 * kh;
                pdst[row * 288 + colbase + lc] = acc[i] + bval;
            }
        }
    }
    __syncthreads();

    for (int idx = tid; idx < 16 * Hh; idx += 128) {
        int n = idx / Hh, hh = idx % Hh;
        float R[9];
#pragma unroll
        for (int j = 0; j < 9; ++j) R[j] = rot[(size_t)(nbase + n) * 9 + j];
        float skv = 0.f;
#pragma unroll
        for (int p = 0; p < Pp; ++p) {
            float qx = pts[n * 288 + hh * 12 + p * 3 + 0];
            float qy = pts[n * 288 + hh * 12 + p * 3 + 1];
            float qz = pts[n * 288 + hh * 12 + p * 3 + 2];
            float kx = pts[n * 288 + 144 + hh * 12 + p * 3 + 0];
            float ky = pts[n * 288 + 144 + hh * 12 + p * 3 + 1];
            float kz = pts[n * 288 + 144 + hh * 12 + p * 3 + 2];
#pragma unroll
            for (int j = 0; j < 3; ++j) {
                float qg = qx * R[0 * 3 + j] + qy * R[1 * 3 + j] + qz * R[2 * 3 + j];
                float kg = kx * R[0 * 3 + j] + ky * R[1 * 3 + j] + kz * R[2 * 3 + j];
                qgb[(size_t)(nbase + n) * PD + hh * 12 + p * 3 + j] = f2bf(qg);
                kgb[(size_t)(nbase + n) * PD + hh * 12 + p * 3 + j] = f2bf(kg);
                skv += kg * kg;
            }
        }
        skw[hh * Nres + nbase + n] = skv;
    }
}

// ---------------- Kernel B: fused flash attention (split-K) ----------------
// grid (64 qtiles, NSPLIT), block 384 = 12 waves (wave w = head w; waves 0..7
// also own query pairs for the pair-bias GEMM). Streams `pair` exactly once
// with NT hints; k/v/kg/sk tiles double-buffered through async-to-LDS.
#define STAGEB (32 * Cdim * 2 * 2 + 32 * PD * 2 + Hh * 32 * 4)  // 59904 B

__global__ __launch_bounds__(384) void ipa_attn(
    const float* __restrict__ pair, const float* __restrict__ Wpair,
    const __bf16* __restrict__ qb, const __bf16* __restrict__ kb,
    const __bf16* __restrict__ vb, const __bf16* __restrict__ qgb,
    const __bf16* __restrict__ kgb, const float* __restrict__ skw,
    float* __restrict__ Opart, float* __restrict__ Mpart, float* __restrict__ Lpart)
{
    extern __shared__ char smem[];
    // two stage sets (double buffer), then biasT, then Pbuf
    float*  biasT = (float*)(smem + 2 * STAGEB);          // 16 x 32 x 12 f32
    __bf16* Pbuf  = (__bf16*)((char*)biasT + 16 * 32 * Hh * 4); // 12 x 16 x 32

    const int tid = threadIdx.x, lane = tid & 31, w = tid >> 5;
    const int kh = lane >> 4, lc = lane & 15;
    const int nbase = blockIdx.x * 16;
    const int split = blockIdx.y;
    const int h = w;

    // issue one stage set's async copies into stage buffer `b`
    auto stage_issue = [&](int mbase, int b) {
        const unsigned base = (unsigned)(uintptr_t)smem + (unsigned)(b * STAGEB);
        const unsigned kbo = base, vbo = base + 24576, kgo = base + 49152,
                       sko = base + 58368;
        const unsigned kvbase = (unsigned)mbase * (Cdim * 2);      // bytes
        for (int i = tid; i < (32 * Cdim * 2) / 16; i += 384) {    // 1536 x b128
            async_b128(kbo + i * 16, kvbase + i * 16, kb);
            async_b128(vbo + i * 16, kvbase + i * 16, vb);
        }
        const unsigned kgbase = (unsigned)mbase * (PD * 2);
        for (int i = tid; i < (32 * PD * 2) / 16; i += 384)        // 576 x b128
            async_b128(kgo + i * 16, kgbase + i * 16, kgb);
        if (tid < (Hh * 32 * 4) / 16) {                            // 96 x b128
            int hh = tid / 8, j = tid % 8;
            async_b128(sko + tid * 16,
                       (unsigned)((hh * Nres + mbase) * 4 + j * 16), skw);
        }
    };

    // Wpair B-fragments (K=128 -> 4 chunks), head column padded to 16
    v16bf wp[4];
#pragma unroll
    for (int kc = 0; kc < 4; ++kc) {
        F16x16 f;
#pragma unroll
        for (int j = 0; j < 8; ++j) {
            int K = 2 * j + 16 * kh + 32 * kc;
            float lo = (lc < Hh) ? Wpair[K * Hh + lc] : 0.f;
            float hi = (lc < Hh) ? Wpair[(K + 1) * Hh + lc] : 0.f;
            f.u[j] = pack2(lo, hi);
        }
        wp[kc] = f.v;
    }

    // per-head query fragments: contiguous bf16 runs
    F16x16 qf, qg;
    {
        const __bf16* qrow = qb + (size_t)(nbase + lc) * Cdim + h * CH;
        __builtin_memcpy(&qf.u[0], qrow + 16 * kh, 16);
        __builtin_memcpy(&qf.u[4], qrow + 16 * kh + 8, 16);
#pragma unroll
        for (int j = 0; j < 8; ++j) qg.u[j] = 0u;
        const __bf16* qgrow = qgb + (size_t)(nbase + lc) * PD + h * 12;
        if (kh == 0) __builtin_memcpy(&qg.u[0], qgrow, 24);       // K=0..11
    }

    float mrow[8], lrow[8];
    v8f O0 = {}, O1 = {};
#pragma unroll
    for (int i = 0; i < 8; ++i) { mrow[i] = -1e30f; lrow[i] = 0.f; }

    const int MT = (Nres / NSPLIT) / 32;
    stage_issue(split * (Nres / NSPLIT), 0);              // prologue: tile 0

    for (int mt = 0; mt < MT; ++mt) {
        const int mbase = split * (Nres / NSPLIT) + mt * 32;
        char* cur = smem + (mt & 1) * STAGEB;
        __bf16* kbt = (__bf16*)cur;
        __bf16* vbt = (__bf16*)(cur + 24576);
        __bf16* kgt = (__bf16*)(cur + 49152);
        float*  skt = (float*)(cur + 58368);

        asm volatile("s_wait_asynccnt 0" ::: "memory");   // tile mt landed
        __syncthreads();                                  // all waves past mt-1
        if (mt + 1 < MT) stage_issue(mbase + 32, (mt + 1) & 1);  // overlap fetch

        // ---- pair bias: waves 0..7, each owns queries 2w,2w+1 ----
        if (w < 8) {
            for (int qq = 0; qq < 2; ++qq) {
                int qn = 2 * w + qq;
                const float* prow = pair + ((size_t)(nbase + qn) * Nres + mbase) * DPAIR;
                if (mt + 1 < MT) __builtin_prefetch(prow + 32 * DPAIR, 0, 1);
                for (int mh = 0; mh < 2; ++mh) {
                    v8f bacc = {};
                    int mr = lc + 16 * mh;
#pragma unroll
                    for (int kc = 0; kc < 4; ++kc) {
                        v16bf af = a_frag_f32_nt(prow + (size_t)mr * DPAIR + 32 * kc, kh);
                        bacc = wmma_bf16(af, wp[kc], bacc);
                    }
                    if (lc < Hh) {
#pragma unroll
                        for (int i = 0; i < 8; ++i) {
                            int mloc = i + 8 * kh + 16 * mh;
                            biasT[(qn * 32 + mloc) * Hh + lc] = bacc[i];
                        }
                    }
                }
            }
        }
        __syncthreads();

        // ---- per-head logits + online softmax + PV ----
        v8f S[2];
#pragma unroll
        for (int t = 0; t < 2; ++t) {
            int mcol = lc + 16 * t;
            F16x16 kf, kgf;
            __builtin_memcpy(&kf.u[0], &kbt[mcol * Cdim + h * CH + 16 * kh], 32);
#pragma unroll
            for (int j = 0; j < 8; ++j) kgf.u[j] = 0u;
            if (kh == 0)   // K = 0..11 live, rest zero-padded
                __builtin_memcpy(&kgf.u[0], &kgt[mcol * PD + h * 12], 24);
            v8f s = {};  s  = wmma_bf16(qf.v, kf.v, s);
            v8f sp = {}; sp = wmma_bf16(qg.v, kgf.v, sp);
            float skv = skt[h * 32 + mcol];
#pragma unroll
            for (int i = 0; i < 8; ++i) {
                int n = i + 8 * kh;
                float bias = biasT[(n * 32 + mcol) * Hh + h];
                S[t][i] = SCALE * s[i] + bias + SCALE * sp[i] - 0.5f * SCALE * skv;
            }
        }
#pragma unroll
        for (int i = 0; i < 8; ++i) {
            float tmax = fmaxf(S[0][i], S[1][i]);
            for (int mk = 1; mk < 16; mk <<= 1) tmax = fmaxf(tmax, __shfl_xor(tmax, mk));
            float newm = fmaxf(mrow[i], tmax);
            float corr = __expf(mrow[i] - newm);
            float p0 = __expf(S[0][i] - newm);
            float p1 = __expf(S[1][i] - newm);
            float rs = p0 + p1;
            for (int mk = 1; mk < 16; mk <<= 1) rs += __shfl_xor(rs, mk);
            lrow[i] = lrow[i] * corr + rs;
            mrow[i] = newm;
            O0[i] *= corr; O1[i] *= corr;
            S[0][i] = p0; S[1][i] = p1;
        }
        // restage P through LDS into A-operand layout
#pragma unroll
        for (int i = 0; i < 8; ++i) {
            int n = i + 8 * kh;
            Pbuf[w * 512 + n * 32 + lc]      = f2bf(S[0][i]);
            Pbuf[w * 512 + n * 32 + 16 + lc] = f2bf(S[1][i]);
        }
        asm volatile("s_wait_dscnt 0" ::: "memory");
        F16x16 pf;
        __builtin_memcpy(&pf.u[0], &Pbuf[w * 512 + lc * 32 + 16 * kh], 16);
        __builtin_memcpy(&pf.u[4], &Pbuf[w * 512 + lc * 32 + 16 * kh + 8], 16);
#pragma unroll
        for (int t = 0; t < 2; ++t) {
            F16x16 vf;
#pragma unroll
            for (int e = 0; e < 16; ++e)
                vf.v[e] = vbt[(e + 16 * kh) * Cdim + h * CH + 16 * t + lc];
            if (t == 0) O0 = wmma_bf16(pf.v, vf.v, O0);
            else        O1 = wmma_bf16(pf.v, vf.v, O1);
        }
    }

    // ---- write partial (unnormalized) output + softmax stats ----
#pragma unroll
    for (int i = 0; i < 8; ++i) {
        int n = i + 8 * kh;
        Opart[((size_t)split * Nres + nbase + n) * Cdim + h * CH + lc]      = O0[i];
        Opart[((size_t)split * Nres + nbase + n) * Cdim + h * CH + 16 + lc] = O1[i];
        if (lc == 0) {
            Mpart[(split * Hh + h) * Nres + nbase + n] = mrow[i];
            Lpart[(split * Hh + h) * Nres + nbase + n] = lrow[i];
        }
    }
}

// ---------------- Kernel B2: merge split-K partials ----------------
__global__ __launch_bounds__(256) void ipa_combine(
    const float* __restrict__ Opart, const float* __restrict__ Mpart,
    const float* __restrict__ Lpart, float* __restrict__ weighted)
{
    int id = blockIdx.x * blockDim.x + threadIdx.x;
    if (id >= Nres * Cdim) return;
    int n = id / Cdim, c = id % Cdim, h = c / CH;
    float M = -1e30f;
#pragma unroll
    for (int s = 0; s < NSPLIT; ++s) M = fmaxf(M, Mpart[(s * Hh + h) * Nres + n]);
    float L = 0.f, acc = 0.f;
#pragma unroll
    for (int s = 0; s < NSPLIT; ++s) {
        float e = __expf(Mpart[(s * Hh + h) * Nres + n] - M);
        L   += Lpart[(s * Hh + h) * Nres + n] * e;
        acc += Opart[((size_t)s * Nres + n) * Cdim + c] * e;
    }
    weighted[id] = acc / L;
}

// ---------------- Kernel C: output projection + residual + LayerNorm ----------------
__global__ __launch_bounds__(128) void ipa_out(
    const float* __restrict__ weighted, const float* __restrict__ Wout,
    const float* __restrict__ bout, const float* __restrict__ single,
    const float* __restrict__ gamma, const float* __restrict__ beta,
    float* __restrict__ out)
{
    __shared__ float sW[16 * Cdim];
    __shared__ float sx[16 * Cdim];
    const int tid = threadIdx.x, lane = tid & 31, w = tid >> 5;
    const int kh = lane >> 4, lc = lane & 15;
    const int nbase = blockIdx.x * 16;

    for (int i = tid; i < 16 * Cdim; i += 128)
        sW[i] = weighted[(size_t)nbase * Cdim + i];
    __syncthreads();

    for (int ct = w; ct < 24; ct += 4) {
        int colbase = ct * 16;
        v8f acc = {};
        for (int ks = 0; ks < 12; ++ks) {
            v16bf af = a_frag_f32(&sW[lc * Cdim + 32 * ks], kh);
            F16x16 bfm;
#pragma unroll
            for (int j = 0; j < 8; ++j) {
                int Kb = 2 * j + 16 * kh + 32 * ks;
                float lo = Wout[(size_t)Kb * Cdim + colbase + lc];
                float hi = Wout[(size_t)(Kb + 1) * Cdim + colbase + lc];
                bfm.u[j] = pack2(lo, hi);
            }
            acc = wmma_bf16(af, bfm.v, acc);
        }
#pragma unroll
        for (int i = 0; i < 8; ++i) {
            int row = i + 8 * kh, col = colbase + lc;
            sx[row * Cdim + col] = acc[i] + bout[col] +
                                   single[(size_t)(nbase + row) * Cdim + col];
        }
    }
    __syncthreads();

    int row = tid >> 3, sub = tid & 7;
    float sum = 0.f;
    for (int c = sub; c < Cdim; c += 8) sum += sx[row * Cdim + c];
    for (int mk = 1; mk < 8; mk <<= 1) sum += __shfl_xor(sum, mk);
    float mu = sum / (float)Cdim;
    float var = 0.f;
    for (int c = sub; c < Cdim; c += 8) { float d = sx[row * Cdim + c] - mu; var += d * d; }
    for (int mk = 1; mk < 8; mk <<= 1) var += __shfl_xor(var, mk);
    float rstd = rsqrtf(var / (float)Cdim + 1e-5f);
    for (int c = sub; c < Cdim; c += 8)
        out[(size_t)(nbase + row) * Cdim + c] =
            (sx[row * Cdim + c] - mu) * rstd * gamma[c] + beta[c];
}

// ---------------- Launch ----------------
extern "C" void kernel_launch(void* const* d_in, const int* in_sizes, int n_in,
                              void* d_out, int out_size, void* d_ws, size_t ws_size,
                              hipStream_t stream) {
    const float* single = (const float*)d_in[0];
    const float* pair   = (const float*)d_in[1];
    const float* rot    = (const float*)d_in[2];
    // d_in[3] translations: unused by the reference
    const float* Wq = (const float*)d_in[4],  *bq = (const float*)d_in[5];
    const float* Wk = (const float*)d_in[6],  *bk = (const float*)d_in[7];
    const float* Wv = (const float*)d_in[8],  *bv = (const float*)d_in[9];
    const float* Wpair = (const float*)d_in[10];            // bpair (11) cancels in softmax
    const float* Wpq = (const float*)d_in[12], *bpq = (const float*)d_in[13];
    const float* Wpk = (const float*)d_in[14], *bpk = (const float*)d_in[15];
    const float* Wout = (const float*)d_in[16], *bout = (const float*)d_in[17];
    const float* gamma = (const float*)d_in[18], *beta = (const float*)d_in[19];
    float* out = (float*)d_out;

    char* ws = (char*)d_ws;
    __bf16* qb  = (__bf16*)(ws);                         // 1024*384 bf16
    __bf16* kb  = (__bf16*)(ws + 786432);
    __bf16* vb  = (__bf16*)(ws + 1572864);
    __bf16* qgb = (__bf16*)(ws + 2359296);               // 1024*144 bf16
    __bf16* kgb = (__bf16*)(ws + 2654208);
    float*  skw = (float*)(ws + 2949120);                // 12*1024 f32
    float*  Opart = (float*)(ws + 2998272);              // NSPLIT*1024*384 f32
    float*  Mpart = (float*)(ws + 9289728);              // NSPLIT*12*1024 f32
    float*  Lpart = (float*)(ws + 9486336);
    float*  weighted = (float*)(ws + 9682944);           // 1024*384 f32

    ipa_proj<<<64, 128, 0, stream>>>(single, rot, Wq, bq, Wk, bk, Wv, bv,
                                     Wpq, bpq, Wpk, bpk, qb, kb, vb, qgb, kgb, skw);

    size_t shmem = 2 * STAGEB                 // double-buffered k/v/kg/sk
                 + 16 * 32 * Hh * 4           // biasT
                 + Hh * 512 * 2;              // Pbuf
    ipa_attn<<<dim3(64, NSPLIT), 384, shmem, stream>>>(
        pair, Wpair, qb, kb, vb, qgb, kgb, skw, Opart, Mpart, Lpart);

    ipa_combine<<<(Nres * Cdim + 255) / 256, 256, 0, stream>>>(Opart, Mpart, Lpart, weighted);

    ipa_out<<<64, 128, 0, stream>>>(weighted, Wout, bout, single, gamma, beta, out);
}
